// MyUnidirectionalRNN_1571958030773
// MI455X (gfx1250) — matrix operand order repaired and checked by reading, not compiled, not run on hardware
//
#include <hip/hip_runtime.h>

// Problem constants (match reference)
#define BATCH 32
#define SEQ   2048
#define FEATS 128
#define RNN   512
#define NCH   64          // chunks over time
#define CLEN  32          // SEQ / NCH
#define HROW  520         // padded LDS row length (bf16 elems); 1040B = 16B multiple, rotates banks

typedef __attribute__((ext_vector_type(16))) __bf16 v16bf;
typedef __attribute__((ext_vector_type(2)))  __bf16 v2bf;
typedef __attribute__((ext_vector_type(2)))  float  v2f;
typedef __attribute__((ext_vector_type(8)))  float  v8f;

union V16U { v16bf v; uint4 q[2]; unsigned int w[8]; };

// f32 -> bf16 via the compiler's native lowering (RNE); packed where possible.
__device__ __forceinline__ unsigned int pk_bf16(float lo, float hi) {
  v2f f; f[0] = lo; f[1] = hi;
  v2bf p = __builtin_convertvector(f, v2bf);
  return __builtin_bit_cast(unsigned int, p);
}
__device__ __forceinline__ unsigned short bf1(float f) {
  __bf16 h = (__bf16)f;
  return __builtin_bit_cast(unsigned short, h);
}

__device__ __forceinline__ v8f wmma_bf16(v16bf a, v16bf b, v8f c) {
  // (neg_a, A, neg_b, B, c_mod, C, reuse_a, reuse_b)
  return __builtin_amdgcn_wmma_f32_16x16x32_bf16(false, a, false, b, (short)0, c, false, false);
}

// A-tile 16x32 bf16 from row-major bf16 storage (LDS or global), ld in elements.
// ISA layout: lane L holds row M=L%16; h=L/16: v0..3 -> K=2v+8h, v4..7 -> K=16+2(v-4)+8h.
__device__ __forceinline__ v16bf load_A(const unsigned short* base, int ld, int mtile,
                                        int kbase, int lane) {
  int m = mtile * 16 + (lane & 15);
  int h = (lane >> 4) & 1;
  const unsigned short* p = base + m * ld + kbase + 8 * h;
  V16U u;
  u.q[0] = *(const uint4*)(p);        // K = kbase+8h .. +7   (pairs v0..3)
  u.q[1] = *(const uint4*)(p + 16);   // K = kbase+16+8h .. +7 (pairs v4..7)
  return u.v;
}

// B-tile 32x16 bf16 where storage is Bt[n][k] row-major (B[k][n] = Bt[n][k]).
// ISA layout: lane L holds N=L%16; h=L/16: elems 0..7 -> K=kbase+8h+s, elems 8..15 -> K=kbase+16+8h+(s-8).
__device__ __forceinline__ v16bf load_Bt(const unsigned short* base, int ld, int ntile,
                                         int kbase, int lane) {
  int n = ntile * 16 + (lane & 15);
  int kh = (lane >> 4) & 1;
  const unsigned short* p = base + n * ld + kbase + 8 * kh;
  V16U u;
  u.q[0] = *(const uint4*)(p);
  u.q[1] = *(const uint4*)(p + 16);
  return u.v;
}

__device__ __forceinline__ int c_row(int lane, int v) { return (((lane >> 4) & 1) << 3) + v; }

// ---- fp32 -> bf16 conversion (optionally also transposed copy) -------------
__global__ void k_cvt(const float* __restrict__ src, unsigned short* __restrict__ dst,
                      unsigned short* __restrict__ dstT, int rows, int cols) {
  int i = blockIdx.x * blockDim.x + threadIdx.x;
  if (i >= rows * cols) return;
  int r = i / cols, c = i - r * cols;
  unsigned short b = bf1(src[i]);
  dst[i] = b;
  if (dstT) dstT[c * rows + r] = b;
}

// ---- g = W x + b  (each wave: 1 M-tile row, 4 N-tiles share the A-tile) ----
__global__ void k_proj(const unsigned short* __restrict__ xbf, const unsigned short* __restrict__ Wbf,
                       const float* __restrict__ bias, float* __restrict__ out) {
  int lane = threadIdx.x & 31;
  int wave = threadIdx.x >> 5;          // 8 waves/block
  int mt = blockIdx.x;                  // B*T/16 = 4096 M-tiles
  int ntb = wave * 4;                   // 4 N-tiles per wave -> 32 N-tiles/block
  float bv = bias[0];
  v8f acc[4];
#pragma unroll
  for (int j = 0; j < 4; ++j) {
    v8f c = {bv, bv, bv, bv, bv, bv, bv, bv};
    acc[j] = c;
  }
#pragma unroll
  for (int k = 0; k < FEATS; k += 32) {
    v16bf a = load_A(xbf, FEATS, mt, k, lane);     // loaded once, used 4x
#pragma unroll
    for (int j = 0; j < 4; ++j)
      acc[j] = wmma_bf16(a, load_Bt(Wbf, FEATS, ntb + j, k, lane), acc[j]);
  }
#pragma unroll
  for (int j = 0; j < 4; ++j) {
    int col = (ntb + j) * 16 + (lane & 15);
#pragma unroll
    for (int v = 0; v < 8; ++v) {
      int row = mt * 16 + c_row(lane, v);          // flat bt index
      out[row * RNN + col] = acc[j][v];
    }
  }
}

// ---- chunk-local scan: h_t = g_t + h_{t-1}·U^T (h in double-buffered LDS) --
__global__ void k_scan_local(float* __restrict__ out, const unsigned short* __restrict__ Ubf,
                             float* __restrict__ bound) {
  extern __shared__ unsigned short hs[];          // 2 x [BATCH][HROW] bf16
  int lane = threadIdx.x & 31, wave = threadIdx.x >> 5;
  int ch = blockIdx.x;
  int mt = wave >> 3;                   // waves 0-7 -> M-tile 0, 8-15 -> M-tile 1
  int ntb = (wave & 7) * 4;             // 4 N-tiles per wave
  for (int i = threadIdx.x; i < BATCH * HROW; i += blockDim.x) hs[i] = 0;  // h_{-1}=0
  __syncthreads();
  for (int t = 0; t < CLEN; ++t) {
    int cur = t & 1, nxt = cur ^ 1;
    int tt = ch * CLEN + t;
    __builtin_prefetch(&Ubf[(ntb * 16 + lane) * RNN], 0, 1);
    v8f acc[4];
#pragma unroll
    for (int j = 0; j < 4; ++j) {
      int col = (ntb + j) * 16 + (lane & 15);
#pragma unroll
      for (int v = 0; v < 8; ++v) {
        int row = mt * 16 + c_row(lane, v);       // batch index
        acc[j][v] = out[(row * SEQ + tt) * RNN + col]; // C init = g_t
      }
    }
#pragma unroll
    for (int k = 0; k < RNN; k += 32) {
      v16bf a = load_A(hs + cur * BATCH * HROW, HROW, mt, k, lane);
#pragma unroll
      for (int j = 0; j < 4; ++j)
        acc[j] = wmma_bf16(a, load_Bt(Ubf, RNN, ntb + j, k, lane), acc[j]);
    }
#pragma unroll
    for (int j = 0; j < 4; ++j) {
      int col = (ntb + j) * 16 + (lane & 15);
#pragma unroll
      for (int v = 0; v < 8; ++v) {
        int row = mt * 16 + c_row(lane, v);
        out[(row * SEQ + tt) * RNN + col] = acc[j][v];
        hs[nxt * BATCH * HROW + row * HROW + col] = bf1(acc[j][v]);
        if (t == CLEN - 1) bound[(ch * BATCH + row) * RNN + col] = acc[j][v];
      }
    }
    __syncthreads();
  }
}

// ---- matrix squaring Q = P·P (writes Q and Q^T in bf16) --------------------
__global__ void k_sq(const unsigned short* __restrict__ P, const unsigned short* __restrict__ Pt,
                     unsigned short* __restrict__ Q, unsigned short* __restrict__ Qt) {
  int lane = threadIdx.x & 31, wave = threadIdx.x >> 5;
  int mt = blockIdx.x;                  // 32 M-tiles
  int ntb = wave * 4;                   // 8 waves x 4 N-tiles
  v8f acc[4];
#pragma unroll
  for (int j = 0; j < 4; ++j) {
    v8f z = {0.f, 0.f, 0.f, 0.f, 0.f, 0.f, 0.f, 0.f};
    acc[j] = z;
  }
#pragma unroll
  for (int k = 0; k < RNN; k += 32) {
    v16bf a = load_A(P, RNN, mt, k, lane);
#pragma unroll
    for (int j = 0; j < 4; ++j)
      acc[j] = wmma_bf16(a, load_Bt(Pt, RNN, ntb + j, k, lane), acc[j]);
  }
#pragma unroll
  for (int j = 0; j < 4; ++j) {
    int col = (ntb + j) * 16 + (lane & 15);
#pragma unroll
    for (int v = 0; v < 8; ++v) {
      int row = mt * 16 + c_row(lane, v);
      unsigned short bb = bf1(acc[j][v]);
      Q[row * RNN + col]  = bb;
      Qt[col * RNN + row] = bb;
    }
  }
}

// ---- boundary scan (serial over 64 chunks, single workgroup) ---------------
__global__ void k_scan_bound(const float* __restrict__ bound, const unsigned short* __restrict__ P32,
                             float* __restrict__ te) {
  extern __shared__ unsigned short hs[];
  int lane = threadIdx.x & 31, wave = threadIdx.x >> 5;
  int mt = wave >> 3;
  int ntb = (wave & 7) * 4;
  for (int i = threadIdx.x; i < BATCH * HROW; i += blockDim.x) hs[i] = 0;
  __syncthreads();
  for (int c0 = 0; c0 < NCH; ++c0) {
    int cur = c0 & 1, nxt = cur ^ 1;
    v8f acc[4];
#pragma unroll
    for (int j = 0; j < 4; ++j) {
      int col = (ntb + j) * 16 + (lane & 15);
#pragma unroll
      for (int v = 0; v < 8; ++v) {
        int row = mt * 16 + c_row(lane, v);
        acc[j][v] = bound[(c0 * BATCH + row) * RNN + col];
      }
    }
#pragma unroll
    for (int k = 0; k < RNN; k += 32) {
      v16bf a = load_A(hs + cur * BATCH * HROW, HROW, mt, k, lane);
#pragma unroll
      for (int j = 0; j < 4; ++j)
        acc[j] = wmma_bf16(a, load_Bt(P32, RNN, ntb + j, k, lane), acc[j]); // te·(U^CLEN)^T
    }
#pragma unroll
    for (int j = 0; j < 4; ++j) {
      int col = (ntb + j) * 16 + (lane & 15);
#pragma unroll
      for (int v = 0; v < 8; ++v) {
        int row = mt * 16 + c_row(lane, v);
        te[(c0 * BATCH + row) * RNN + col] = acc[j][v];
        hs[nxt * BATCH * HROW + row * HROW + col] = bf1(acc[j][v]);
      }
    }
    __syncthreads();
  }
}

// ---- fix-up: add U^{i+1}·te[c-1] to every step of chunk c ------------------
__global__ void k_fixup(float* __restrict__ out, const unsigned short* __restrict__ Ubf,
                        const float* __restrict__ te) {
  extern __shared__ unsigned short hs[];
  int lane = threadIdx.x & 31, wave = threadIdx.x >> 5;
  int ch = blockIdx.x + 1;                        // chunks 1..NCH-1
  int mt = wave >> 3;
  int ntb = (wave & 7) * 4;
  for (int i = threadIdx.x; i < BATCH * RNN; i += blockDim.x) {
    int r = i >> 9, cc = i & (RNN - 1);
    hs[r * HROW + cc] = bf1(te[((ch - 1) * BATCH + r) * RNN + cc]);
  }
  __syncthreads();
  for (int t = 0; t < CLEN; ++t) {
    int cur = t & 1, nxt = cur ^ 1;
    int tt = ch * CLEN + t;
    __builtin_prefetch(&Ubf[(ntb * 16 + lane) * RNN], 0, 1);
    v8f acc[4];
#pragma unroll
    for (int j = 0; j < 4; ++j) {
      v8f z = {0.f, 0.f, 0.f, 0.f, 0.f, 0.f, 0.f, 0.f};
      acc[j] = z;
    }
#pragma unroll
    for (int k = 0; k < RNN; k += 32) {
      v16bf a = load_A(hs + cur * BATCH * HROW, HROW, mt, k, lane);
#pragma unroll
      for (int j = 0; j < 4; ++j)
        acc[j] = wmma_bf16(a, load_Bt(Ubf, RNN, ntb + j, k, lane), acc[j]);
    }
#pragma unroll
    for (int j = 0; j < 4; ++j) {
      int col = (ntb + j) * 16 + (lane & 15);
#pragma unroll
      for (int v = 0; v < 8; ++v) {
        int row = mt * 16 + c_row(lane, v);
        int gi = (row * SEQ + tt) * RNN + col;
        out[gi] += acc[j][v];
        hs[nxt * BATCH * HROW + row * HROW + col] = bf1(acc[j][v]);
      }
    }
    __syncthreads();
  }
}

extern "C" void kernel_launch(void* const* d_in, const int* in_sizes, int n_in,
                              void* d_out, int out_size, void* d_ws, size_t ws_size,
                              hipStream_t stream) {
  const float* x = (const float*)d_in[0];   // [B,T,F]
  const float* W = (const float*)d_in[1];   // [RNN,F]
  const float* U = (const float*)d_in[2];   // [RNN,RNN]
  const float* b = (const float*)d_in[3];   // [1]
  float* out = (float*)d_out;               // [B,T,RNN]

  char* ws = (char*)d_ws;                   // ~28.4 MB used
  unsigned short* Wbf  = (unsigned short*)(ws + 0);        // 512*128*2
  unsigned short* Ubf  = (unsigned short*)(ws + 131072);   // 512*512*2
  unsigned short* Utbf = (unsigned short*)(ws + 655360);
  unsigned short* Pa   = (unsigned short*)(ws + 1179648);
  unsigned short* Pta  = (unsigned short*)(ws + 1703936);
  unsigned short* Pb   = (unsigned short*)(ws + 2228224);
  unsigned short* Ptb  = (unsigned short*)(ws + 2752512);
  float* bound = (float*)(ws + 3276800);                   // [NCH][B][RNN]
  float* te    = (float*)(ws + 7471104);                   // [NCH][B][RNN]
  unsigned short* xbf  = (unsigned short*)(ws + 11665408); // [B*T][F] bf16, 16.8 MB

  size_t lds = (size_t)2 * BATCH * HROW * sizeof(unsigned short); // 66,560 B dynamic LDS

  k_cvt<<<(BATCH * SEQ * FEATS + 255) / 256, 256, 0, stream>>>(x, xbf, (unsigned short*)nullptr,
                                                               BATCH * SEQ, FEATS);
  k_cvt<<<(RNN * RNN + 255) / 256, 256, 0, stream>>>(U, Ubf, Utbf, RNN, RNN);
  k_cvt<<<(RNN * FEATS + 255) / 256, 256, 0, stream>>>(W, Wbf, (unsigned short*)nullptr, RNN, FEATS);
  k_proj<<<BATCH * SEQ / 16, 256, 0, stream>>>(xbf, Wbf, b, out);
  k_scan_local<<<NCH, 512, lds, stream>>>(out, Ubf, bound);
  k_sq<<<RNN / 16, 256, 0, stream>>>(Ubf, Utbf, Pa, Pta);   // U^2
  k_sq<<<RNN / 16, 256, 0, stream>>>(Pa, Pta, Pb, Ptb);     // U^4
  k_sq<<<RNN / 16, 256, 0, stream>>>(Pb, Ptb, Pa, Pta);     // U^8
  k_sq<<<RNN / 16, 256, 0, stream>>>(Pa, Pta, Pb, Ptb);     // U^16
  k_sq<<<RNN / 16, 256, 0, stream>>>(Pb, Ptb, Pa, Pta);     // U^32 -> Pa
  k_scan_bound<<<1, 512, lds, stream>>>(bound, Pa, te);
  k_fixup<<<NCH - 1, 512, lds, stream>>>(out, Ubf, te);
}
